// EncoderSwin_10325101380010
// MI455X (gfx1250) — matrix operand order repaired and checked by reading, not compiled, not run on hardware
//
#include <hip/hip_runtime.h>

// ---------------------------------------------------------------------------
// CDNA5 (gfx1250) Swin forward. All matmuls via v_wmma_f32_16x16x32_bf16.
// wave32; A/B k order: ISA 7.12.2 in attention, contiguous (16h+s) in gemm_bf16. All WMMA operands are
// staged in bf16 (weights pre-converted once per launch; activation producers
// emit bf16 directly), so the GEMM inner loop is loads + WMMA only.
// ---------------------------------------------------------------------------

typedef float v8f   __attribute__((ext_vector_type(8)));
typedef __bf16 v16bf __attribute__((ext_vector_type(16)));

union BF16x16 {
  v16bf v;
  unsigned short u[16];
  uint4 q[2];          // two 16-byte halves, memory-order == element-order
};

__device__ __forceinline__ unsigned short f2bf(float f) {
  unsigned int x = __float_as_uint(f);
  x += 0x7fffu + ((x >> 16) & 1u);          // round-to-nearest-even truncate
  return (unsigned short)(x >> 16);
}

__device__ __forceinline__ uint4 ldq(const unsigned short* p) {
  return *reinterpret_cast<const uint4*>(p);
}

__device__ __forceinline__ v8f wmma_bf16(const BF16x16& a, const BF16x16& b, v8f c) {
  return __builtin_amdgcn_wmma_f32_16x16x32_bf16(false, a.v, false, b.v, (short)0, c, false, false);
}

// token index for window w=(bb,wy,wx), in-window position p (0..48), roll(shift)
__device__ __forceinline__ int map_tok(int bb, int wy, int wx, int p,
                                       int H, int W, int shift) {
  int py = p / 7, px = p % 7;
  int y = wy * 7 + py + shift; if (y >= H) y -= H;
  int x = wx * 7 + px + shift; if (x >= W) x -= W;
  return (bb * H + y) * W + x;
}

__device__ __forceinline__ int regid(int c, int n, int shift) {
  return (c < n - 7) ? 0 : ((c < n - shift) ? 1 : 2);
}

// ---------------------------------------------------------------------------
// f32 -> bf16 elementwise (weight staging, head input staging)
// ---------------------------------------------------------------------------
__global__ void f32_to_bf16_k(const float* __restrict__ in,
                              unsigned short* __restrict__ out, int n) {
  const int i = blockIdx.x * 256 + threadIdx.x;
  if (i < n) out[i] = f2bf(in[i]);
}

// ---------------------------------------------------------------------------
// D = A(MxK,bf16) @ Wt(N,K,bf16)^T [+bias] [+res] [gelu]; fp32 accum.
// Optional f32 and/or bf16 outputs. M%16==0, N%32==0, K%32==0.
// Block=256 (8 waves), wave tile 16x32 (2 WMMAs / K-step).
// Inner loop: 6x b128 loads + 2 WMMA, no conversion VALU.
// ---------------------------------------------------------------------------
__global__ __launch_bounds__(256) void gemm_bf16(
    const unsigned short* __restrict__ A, int lda,
    const unsigned short* __restrict__ Wt,
    const float* __restrict__ bias,
    const float* __restrict__ res,
    float* __restrict__ D,
    unsigned short* __restrict__ Dh,
    int ldd, int M, int N, int K, int act)
{
  const int wave = threadIdx.x >> 5;
  const int lane = threadIdx.x & 31;
  const int hl   = lane >> 4;
  const int l15  = lane & 15;
  const int m0 = (blockIdx.y * 8 + wave) * 16;
  if (m0 >= M) return;               // wave-uniform: EXEC stays full for WMMA
  const int n0 = blockIdx.x * 32;

  const unsigned short* arow = A  + (size_t)(m0 + l15) * lda;
  const unsigned short* wr0  = Wt + (size_t)(n0 + l15) * K;
  const unsigned short* wr1  = Wt + (size_t)(n0 + 16 + l15) * K;
  const int ab = hl * 8;             // unused; A also reads K run [kb,kb+16)
  const int kb = hl * 16;            // B: K run  [kb,kb+16)

  v8f acc0, acc1;
  #pragma unroll
  for (int r = 0; r < 8; ++r) { acc0[r] = 0.f; acc1[r] = 0.f; }

  for (int k0 = 0; k0 < K; k0 += 32) {
    if (k0 + 32 < K) __builtin_prefetch(arow + k0 + 32, 0, 1);
    BF16x16 af, b0, b1;
    af.q[0] = ldq(arow + k0 + kb);
    af.q[1] = ldq(arow + k0 + kb + 8);
    b0.q[0] = ldq(wr0 + k0 + kb);
    b0.q[1] = ldq(wr0 + k0 + kb + 8);
    b1.q[0] = ldq(wr1 + k0 + kb);
    b1.q[1] = ldq(wr1 + k0 + kb + 8);
    acc0 = wmma_bf16(af, b0, acc0);
    acc1 = wmma_bf16(af, b1, acc1);
  }

  #pragma unroll
  for (int r = 0; r < 8; ++r) {
    const int m = m0 + hl * 8 + r;
    #pragma unroll
    for (int nt = 0; nt < 2; ++nt) {
      const int n = n0 + nt * 16 + l15;
      float v = nt ? acc1[r] : acc0[r];
      if (bias) v += bias[n];
      if (res)  v += res[(size_t)m * ldd + n];
      if (act == 1) v = 0.5f * v * (1.0f + erff(v * 0.70710678118654752f));
      if (D)  D[(size_t)m * ldd + n]  = v;
      if (Dh) Dh[(size_t)m * ldd + n] = f2bf(v);
    }
  }
}

// ---------------------------------------------------------------------------
// LayerNorm: one block per row; f32 and/or bf16 outputs; in-place safe for f32.
// ---------------------------------------------------------------------------
__global__ __launch_bounds__(256) void layernorm_k(
    const float* __restrict__ in, const float* __restrict__ g,
    const float* __restrict__ bta, float* __restrict__ outf,
    unsigned short* __restrict__ outh, int C)
{
  __shared__ float sh[256];
  const int row = blockIdx.x;
  const int tid = threadIdx.x;
  const float* x = in + (size_t)row * C;
  float* of = outf ? outf + (size_t)row * C : nullptr;
  unsigned short* oh = outh ? outh + (size_t)row * C : nullptr;
  float s = 0.f;
  for (int c = tid; c < C; c += 256) s += x[c];
  sh[tid] = s; __syncthreads();
  for (int st = 128; st > 0; st >>= 1) { if (tid < st) sh[tid] += sh[tid + st]; __syncthreads(); }
  const float mu = sh[0] / C;
  __syncthreads();
  float v = 0.f;
  for (int c = tid; c < C; c += 256) { float d = x[c] - mu; v += d * d; }
  sh[tid] = v; __syncthreads();
  for (int st = 128; st > 0; st >>= 1) { if (tid < st) sh[tid] += sh[tid + st]; __syncthreads(); }
  const float rs = rsqrtf(sh[0] / C + 1e-5f);
  for (int c = tid; c < C; c += 256) {
    const float o = (x[c] - mu) * rs * g[c] + bta[c];
    if (of) of[c] = o;
    if (oh) oh[c] = f2bf(o);
  }
}

// ---------------------------------------------------------------------------
// Patch embed: 4x4/stride-4 conv (1 input channel) + bias, token-major f32.
// ---------------------------------------------------------------------------
__global__ void patch_embed_k(const float* __restrict__ x, const float* __restrict__ w,
                              const float* __restrict__ bias, float* __restrict__ y,
                              int B, int OH, int OW, int IH, int IW)
{
  int idx = blockIdx.x * 256 + threadIdx.x;
  const int total = B * OH * OW * 96;
  if (idx >= total) return;
  const int c = idx % 96; int t = idx / 96;
  const int ox = t % OW; t /= OW;
  const int oy = t % OH; const int bb = t / OH;
  const float* xp = x + ((size_t)bb * IH + oy * 4) * IW + ox * 4;
  const float* wc = w + c * 16;
  float s = bias[c];
  #pragma unroll
  for (int ky = 0; ky < 4; ++ky)
    #pragma unroll
    for (int kx = 0; kx < 4; ++kx)
      s += wc[ky * 4 + kx] * xp[ky * IW + kx];
  y[idx] = s;
}

// ---------------------------------------------------------------------------
// Attention scores: S[w][h][i][j] = q_i . k_j (unscaled; scale folded into
// softmax). qkv is bf16 token-major [q(C)|k(C)|v(C)]. One block per (window,
// head); 4 waves over i-tiles; 4 WMMAs per wave (j tiles); K=d=32.
// ---------------------------------------------------------------------------
__global__ __launch_bounds__(128) void attn_scores_k(
    const unsigned short* __restrict__ qkv, float* __restrict__ S,
    int C, int nH, int H, int W, int nwy, int nwx, int shift)
{
  const int w = blockIdx.x, h = blockIdx.y;
  const int wave = threadIdx.x >> 5, lane = threadIdx.x & 31;
  const int hl = lane >> 4, l15 = lane & 15;
  const int wx = w % nwx; int t = w / nwx;
  const int wy = t % nwy; const int bb = t / nwy;
  const int i0 = wave * 16;
  const int i  = i0 + l15;

  BF16x16 af;
  #pragma unroll
  for (int e = 0; e < 16; ++e) af.u[e] = 0;
  if (i < 49) {
    const int tok = map_tok(bb, wy, wx, i, H, W, shift);
    const unsigned short* qp = qkv + (size_t)tok * 3 * C + h * 32;
    const int ab = hl * 8;
    af.q[0] = ldq(qp + ab);
    af.q[1] = ldq(qp + ab + 16);
  }

  v8f acc[4];
  #pragma unroll
  for (int jt = 0; jt < 4; ++jt)
    #pragma unroll
    for (int r = 0; r < 8; ++r) acc[jt][r] = 0.f;

  #pragma unroll
  for (int jt = 0; jt < 4; ++jt) {
    const int j = jt * 16 + l15;
    BF16x16 bf;
    #pragma unroll
    for (int e = 0; e < 16; ++e) bf.u[e] = 0;
    if (j < 49) {
      const int tokj = map_tok(bb, wy, wx, j, H, W, shift);
      const unsigned short* kp = qkv + (size_t)tokj * 3 * C + C + h * 32 + hl * 8;
      bf.q[0] = ldq(kp);
      bf.q[1] = ldq(kp + 16);
    }
    acc[jt] = wmma_bf16(af, bf, acc[jt]);
  }

  float* Sw = S + ((size_t)w * nH + h) * 49 * 49;
  #pragma unroll
  for (int r = 0; r < 8; ++r) {
    const int ii = i0 + hl * 8 + r;
    if (ii < 49) {
      #pragma unroll
      for (int jt = 0; jt < 4; ++jt) {
        const int j = jt * 16 + l15;
        if (j < 49) Sw[ii * 49 + j] = acc[jt][r];
      }
    }
  }
}

// ---------------------------------------------------------------------------
// Softmax over last dim (49): v = scale*S + rel_bias + shift_mask, then
// softmax; emits bf16 probabilities. One thread per (window, head, row).
// ---------------------------------------------------------------------------
__global__ void softmax_bias_k(float* __restrict__ S, unsigned short* __restrict__ P,
                               const float* __restrict__ rel, float scale,
                               int nWin, int nH, int nwy, int nwx, int H, int W, int shift)
{
  const int idx = blockIdx.x * 256 + threadIdx.x;
  const int total = nWin * nH * 49;
  if (idx >= total) return;
  const int i = idx % 49; int t = idx / 49;
  const int h = t % nH; const int w = t / nH;
  float* row = S + ((size_t)w * nH + h) * 49 * 49 + (size_t)i * 49;
  unsigned short* pout = P + ((size_t)w * nH + h) * 49 * 49 + (size_t)i * 49;
  const int r1 = i / 7, c1 = i % 7;
  const int wx = w % nwx, wyb = (w / nwx) % nwy;
  int cnt1 = 0;
  if (shift) cnt1 = regid(wyb * 7 + r1, H, shift) * 3 + regid(wx * 7 + c1, W, shift);
  float m = -1e30f;
  for (int j = 0; j < 49; ++j) {
    const int r2 = j / 7, c2 = j % 7;
    float bsv = rel[((r1 - r2 + 6) * 13 + (c1 - c2 + 6)) * nH + h];
    if (shift) {
      const int cnt2 = regid(wyb * 7 + r2, H, shift) * 3 + regid(wx * 7 + c2, W, shift);
      if (cnt2 != cnt1) bsv -= 100.f;
    }
    const float vv = row[j] * scale + bsv;
    row[j] = vv;
    m = fmaxf(m, vv);
  }
  float ssum = 0.f;
  for (int j = 0; j < 49; ++j) { const float e = expf(row[j] - m); row[j] = e; ssum += e; }
  const float inv = 1.f / ssum;
  for (int j = 0; j < 49; ++j) pout[j] = f2bf(row[j] * inv);
}

// ---------------------------------------------------------------------------
// O = probs(bf16) @ V(bf16) per (window, head); M=49->64, N=32, K=49->64 over
// two WMMA K-steps. Output bf16, token-major (performs un-window + roll-back).
// ---------------------------------------------------------------------------
__global__ __launch_bounds__(128) void attn_av_k(
    const unsigned short* __restrict__ P, const unsigned short* __restrict__ qkv,
    unsigned short* __restrict__ O,
    int C, int nH, int H, int W, int nwy, int nwx, int shift)
{
  const int w = blockIdx.x, h = blockIdx.y;
  const int wave = threadIdx.x >> 5, lane = threadIdx.x & 31;
  const int hl = lane >> 4, l15 = lane & 15;
  const int wx = w % nwx; int t = w / nwx;
  const int wy = t % nwy; const int bb = t / nwy;
  const int i0 = wave * 16;
  const int i  = i0 + l15;
  const unsigned short* Pw = P + ((size_t)w * nH + h) * 49 * 49;

  v8f acc0, acc1;
  #pragma unroll
  for (int r = 0; r < 8; ++r) { acc0[r] = 0.f; acc1[r] = 0.f; }

  for (int k0 = 0; k0 < 64; k0 += 32) {
    BF16x16 af;
    #pragma unroll
    for (int e = 0; e < 16; ++e) af.u[e] = 0;
    if (i < 49) {
      const int ab = hl * 8;
      #pragma unroll
      for (int e = 0; e < 8; ++e) {
        const int k = k0 + ab + e;
        if (k < 49) af.u[e] = Pw[i * 49 + k];
      }
      #pragma unroll
      for (int e = 0; e < 8; ++e) {
        const int k = k0 + ab + 16 + e;
        if (k < 49) af.u[8 + e] = Pw[i * 49 + k];
      }
    }
    BF16x16 b0, b1;
    #pragma unroll
    for (int e = 0; e < 16; ++e) { b0.u[e] = 0; b1.u[e] = 0; }
    #pragma unroll
    for (int e = 0; e < 16; ++e) {
      const int k = k0 + hl * 8 + (e < 8 ? e : e + 8);
      if (k < 49) {
        const int tokk = map_tok(bb, wy, wx, k, H, W, shift);
        const unsigned short* vp = qkv + (size_t)tokk * 3 * C + 2 * C + h * 32;
        b0.u[e] = vp[l15];
        b1.u[e] = vp[16 + l15];
      }
    }
    acc0 = wmma_bf16(af, b0, acc0);
    acc1 = wmma_bf16(af, b1, acc1);
  }

  #pragma unroll
  for (int r = 0; r < 8; ++r) {
    const int ii = i0 + hl * 8 + r;
    if (ii < 49) {
      const int tok = map_tok(bb, wy, wx, ii, H, W, shift);
      unsigned short* op = O + (size_t)tok * C + h * 32;
      op[l15]      = f2bf(acc0[r]);
      op[16 + l15] = f2bf(acc1[r]);
    }
  }
}

// ---------------------------------------------------------------------------
// Patch merge gather: (B,H,W,C) f32 -> (B,H/2,W/2,4C) f32, order [x0,x1,x2,x3].
// ---------------------------------------------------------------------------
__global__ void patch_merge_gather_k(const float* __restrict__ y, float* __restrict__ z,
                                     int B, int H, int W, int C)
{
  const int Ho = H / 2, Wo = W / 2, C4 = 4 * C;
  int idx = blockIdx.x * 256 + threadIdx.x;
  const int total = B * Ho * Wo * C4;
  if (idx >= total) return;
  const int cc = idx % C4; int t = idx / C4;
  const int j = t % Wo; t /= Wo;
  const int i = t % Ho; const int bb = t / Ho;
  const int q = cc / C, c = cc % C;
  const int dy = (q == 1 || q == 3) ? 1 : 0;
  const int dx = (q >= 2) ? 1 : 0;
  z[idx] = y[(((size_t)bb * H + 2 * i + dy) * W + 2 * j + dx) * C + c];
}

// ---------------------------------------------------------------------------
// Host orchestration.
// Input flatten order (recursive dict-insertion order of setup_inputs()):
//   0:x 1:patch_w 2:patch_b 3:patch_ln_g 4:patch_ln_b
//   5 + 13*blk + {0:ln1_g 1:ln1_b 2:qkv_w 3:qkv_b 4:proj_w 5:proj_b 6:rel
//                 7:ln2_g 8:ln2_b 9:fc1_w 10:fc1_b 11:fc2_w 12:fc2_b}
//   161 + 3*m + {0:ln_g 1:ln_b 2:red_w}   (3 merges)
//   170:proj_w(head) 171:proj_b(head)
// Workspace regions (bytes):
//   GQ  bf16  77,070,336  qkv(tokens x 3C) / fc1-h(tokens x 4C)
//   SC  f32   59,006,976  attn scores / merge concat
//   PB  bf16  29,503,488  softmax probs
//   TB  bf16  19,267,584  LN out / attn-O / merge-LN out / head input
//   Y   f32   38,535,168  residual stream
//   WB  bf16  55,640,064  converted weights
// ---------------------------------------------------------------------------
extern "C" void kernel_launch(void* const* d_in, const int* in_sizes, int n_in,
                              void* d_out, int out_size, void* d_ws, size_t ws_size,
                              hipStream_t stream)
{
  (void)in_sizes; (void)n_in; (void)out_size; (void)ws_size;
  auto F = [&](int i) { return (const float*)d_in[i]; };

  char* ws = (char*)d_ws;
  const size_t GQ_SZ = (size_t)100352 * 384 * 2;
  const size_t SC_SZ = (size_t)2048 * 3 * 49 * 49 * 4;
  const size_t PB_SZ = (size_t)2048 * 3 * 49 * 49 * 2;
  const size_t TB_SZ = (size_t)100352 * 96 * 2;
  const size_t Y_SZ  = (size_t)100352 * 96 * 4;

  unsigned short* gqh  = (unsigned short*)ws;                                // GQ
  float*          sbuf = (float*)(ws + GQ_SZ);                               // SC
  unsigned short* pbuf = (unsigned short*)(ws + GQ_SZ + SC_SZ);              // PB
  unsigned short* tbh  = (unsigned short*)(ws + GQ_SZ + SC_SZ + PB_SZ);      // TB
  float*          y    = (float*)(ws + GQ_SZ + SC_SZ + PB_SZ + TB_SZ);       // Y
  unsigned short* wbf  = (unsigned short*)(ws + GQ_SZ + SC_SZ + PB_SZ + TB_SZ + Y_SZ);

  const int DEP[4]   = {2, 2, 6, 2};
  const int NHEAD[4] = {3, 6, 12, 24};
  const float SCALE  = 0.17677669529663687f; // 1/sqrt(32)
  const int B = 16;

  // ---- stage all GEMM weights as bf16 (once per launch, stream-ordered) ----
  size_t woff = 0;
  auto CVT = [&](const float* src, size_t n) -> const unsigned short* {
    unsigned short* dst = wbf + woff;
    f32_to_bf16_k<<<((int)n + 255) / 256, 256, 0, stream>>>(src, dst, (int)n);
    woff += n;
    return dst;
  };
  const unsigned short *Wq[12], *Wp[12], *W1[12], *W2[12], *Wr[3], *Wh;
  {
    int Cc = 96, blk = 0;
    for (int s = 0; s < 4; ++s) {
      for (int d = 0; d < DEP[s]; ++d, ++blk) {
        auto P = [&](int f) { return F(5 + 13 * blk + f); };
        Wq[blk] = CVT(P(2),  (size_t)3 * Cc * Cc);
        Wp[blk] = CVT(P(4),  (size_t)Cc * Cc);
        W1[blk] = CVT(P(9),  (size_t)4 * Cc * Cc);
        W2[blk] = CVT(P(11), (size_t)4 * Cc * Cc);
      }
      if (s < 3) { Wr[s] = CVT(F(161 + 3 * s + 2), (size_t)8 * Cc * Cc); Cc *= 2; }
    }
    Wh = CVT(F(170), (size_t)512 * 768);
  }

  auto GEMM = [&](const unsigned short* A, int lda, const unsigned short* Wt,
                  const float* bias, const float* res, float* D, unsigned short* Dh,
                  int M, int N, int K, int act) {
    dim3 g(N / 32, (M + 127) / 128);
    gemm_bf16<<<g, 256, 0, stream>>>(A, lda, Wt, bias, res, D, Dh, N, M, N, K, act);
  };

  // ---- patch embed + LN ----
  int H = 56, W = 112, C = 96;
  int tokens = B * H * W;
  patch_embed_k<<<(tokens * 96 + 255) / 256, 256, 0, stream>>>(
      F(0), F(1), F(2), y, B, H, W, 224, 448);
  layernorm_k<<<tokens, 256, 0, stream>>>(y, F(3), F(4), y, nullptr, C);

  // ---- stages ----
  int blk = 0;
  for (int s = 0; s < 4; ++s) {
    const int nH = NHEAD[s];
    for (int d = 0; d < DEP[s]; ++d, ++blk) {
      auto P = [&](int f) { return F(5 + 13 * blk + f); };
      const int shift = (d & 1) ? 3 : 0;
      const int nwy = H / 7, nwx = W / 7, nWin = B * nwy * nwx;

      // LN1 -> bf16 tbh
      layernorm_k<<<tokens, 256, 0, stream>>>(y, P(0), P(1), nullptr, tbh, C);
      // qkv GEMM -> bf16 gqh (tokens x 3C)
      GEMM(tbh, C, Wq[blk], P(3), nullptr, nullptr, gqh, tokens, 3 * C, C, 0);
      // scores -> f32 sbuf
      attn_scores_k<<<dim3(nWin, nH), 128, 0, stream>>>(
          gqh, sbuf, C, nH, H, W, nwy, nwx, shift);
      // scale + rel bias + mask + softmax -> bf16 pbuf
      { const int rows = nWin * nH * 49;
        softmax_bias_k<<<(rows + 255) / 256, 256, 0, stream>>>(
            sbuf, pbuf, P(6), SCALE, nWin, nH, nwy, nwx, H, W, shift); }
      // probs @ V -> bf16 tbh (token-major, un-windowed)
      attn_av_k<<<dim3(nWin, nH), 128, 0, stream>>>(
          pbuf, gqh, tbh, C, nH, H, W, nwy, nwx, shift);
      // proj GEMM + residual -> f32 y
      GEMM(tbh, C, Wp[blk], P(5), y, y, nullptr, tokens, C, C, 0);
      // LN2 -> bf16 tbh
      layernorm_k<<<tokens, 256, 0, stream>>>(y, P(7), P(8), nullptr, tbh, C);
      // fc1 GEMM + GELU -> bf16 gqh (tokens x 4C)
      GEMM(tbh, C, W1[blk], P(10), nullptr, nullptr, gqh, tokens, 4 * C, C, 1);
      // fc2 GEMM + residual -> f32 y
      GEMM(gqh, 4 * C, W2[blk], P(12), y, y, nullptr, tokens, C, 4 * C, 0);
    }
    if (s < 3) {
      auto Mg = [&](int f) { return F(161 + 3 * s + f); };
      const int to = tokens / 4;
      patch_merge_gather_k<<<(to * 4 * C + 255) / 256, 256, 0, stream>>>(
          y, sbuf, B, H, W, C);
      layernorm_k<<<to, 256, 0, stream>>>(sbuf, Mg(0), Mg(1), nullptr, tbh, 4 * C);
      GEMM(tbh, 4 * C, Wr[s], nullptr, nullptr, y, nullptr, to, 2 * C, 4 * C, 0);
      H /= 2; W /= 2; C *= 2; tokens = to;
    }
  }

  // ---- final head: y(1568x768) -> bf16 -> @ Wh^T + b -> d_out ----
  f32_to_bf16_k<<<(tokens * 768 + 255) / 256, 256, 0, stream>>>(y, tbh, tokens * 768);
  GEMM(tbh, 768, Wh, F(171), nullptr, (float*)d_out, nullptr, tokens, 512, 768, 0);
}
